// AnimDiT_14929306321249
// MI455X (gfx1250) — 16-bit and 8-bit matrix operand order repaired and checked by reading, compiled, not run on hardware
//
#include <hip/hip_runtime.h>
#include <hip/hip_bf16.h>
#include <stdint.h>


typedef __attribute__((ext_vector_type(8)))  int       v8i;
typedef __attribute__((ext_vector_type(8)))  float     v8f;
typedef __attribute__((ext_vector_type(16))) _Float16  v16h;

#define DIMD   768
#define HEADS  12
#define HD     64
#define SEQ    2048
#define BATCH  2
#define MTOT   (BATCH*SEQ)
#define IFULL  4096
#define IHALF  2048
#define NCH    32
#define CLEN   (SEQ/NCH)
#define EPS_BIT 1e-5f
#define EPS_LN  1e-6f

__device__ __forceinline__ float sigm(float x) { return 1.0f / (1.0f + __expf(-x)); }

__device__ __forceinline__ float block_red_sum(float v, float* red) {
    int tid = threadIdx.x;
    red[tid] = v; __syncthreads();
    for (int s = 128; s > 0; s >>= 1) { if (tid < s) red[tid] += red[tid + s]; __syncthreads(); }
    float r = red[0]; __syncthreads();
    return r;
}
__device__ __forceinline__ float block_red_max(float v, float* red) {
    int tid = threadIdx.x;
    red[tid] = v; __syncthreads();
    for (int s = 128; s > 0; s >>= 1) { if (tid < s) red[tid] = fmaxf(red[tid], red[tid + s]); __syncthreads(); }
    float r = red[0]; __syncthreads();
    return r;
}

__global__ void __launch_bounds__(256) abs_sum_kernel(const float* __restrict__ w, int n,
                                                      float* __restrict__ accum) {
    __shared__ float red[256];
    float s = 0.f;
    for (int i = blockIdx.x * 256 + threadIdx.x; i < n; i += gridDim.x * 256) s += fabsf(w[i]);
    float tot = block_red_sum(s, red);
    if (threadIdx.x == 0) atomicAdd(accum, tot);
}

__global__ void __launch_bounds__(256) wquant_kernel(const float* __restrict__ w,
                                                     signed char* __restrict__ q, int n,
                                                     const float* __restrict__ accum,
                                                     float* __restrict__ wsc) {
    int i = blockIdx.x * 256 + threadIdx.x;
    float inv = fmaxf(*accum / (float)n, 1e-5f);
    if (i == 0) *wsc = inv;
    if (i < n) {
        float s = 1.0f / inv;
        float qq = rintf(w[i] * s);
        qq = fminf(fmaxf(qq, -1.0f), 1.0f);
        q[i] = (signed char)qq;
    }
}

__global__ void __launch_bounds__(256) rmsq_kernel(const float* __restrict__ X,
                                                   const float* __restrict__ g,
                                                   signed char* __restrict__ Q,
                                                   float* __restrict__ ascale, int Kd) {
    __shared__ float red[256];
    int m = blockIdx.x, tid = threadIdx.x;
    const float* x = X + (size_t)m * Kd;
    float ss = 0.f;
    for (int k = tid; k < Kd; k += 256) { float v = x[k]; ss += v * v; }
    float tot = block_red_sum(ss, red);
    float r = rsqrtf(tot / (float)Kd + EPS_BIT);
    float amax = 0.f;
    for (int k = tid; k < Kd; k += 256) amax = fmaxf(amax, fabsf(x[k] * r * g[k]));
    float am = block_red_max(amax, red);
    float s127 = 127.0f / fmaxf(am, 1e-5f);
    for (int k = tid; k < Kd; k += 256) {
        float qv = rintf(x[k] * r * g[k] * s127);
        qv = fminf(fmaxf(qv, -128.0f), 127.0f);
        Q[(size_t)m * Kd + k] = (signed char)qv;
    }
    if (tid == 0) ascale[m] = 1.0f / s127;
}

__global__ void __launch_bounds__(256) gemm_iu8_kernel(const signed char* __restrict__ A,
                                                       const signed char* __restrict__ W,
                                                       float* __restrict__ C,
                                                       const float* __restrict__ ascale,
                                                       const float* __restrict__ wscale,
                                                       int M, int N, int Kd) {
    __shared__ __align__(16) signed char btile[2][64 * 32];
    int tid  = threadIdx.x;
    int wave = tid >> 5, lane = tid & 31;
    int nBlkN = N >> 5;
    int blkM = blockIdx.x / nBlkN;
    int blkN = blockIdx.x % nBlkN;
    int tm   = blkM * 8 + wave;
    int row  = lane & 15;
    int half = lane >> 4;

    const signed char* Ap = A + (size_t)(tm * 16 + row) * Kd + half * 8;
    const signed char* Wp = W + (size_t)blkN * 32;

    int ldrow = tid >> 1;
    int ldcol = (tid & 1) << 4;
    unsigned lds0 = (unsigned)(uintptr_t)&btile[0][0];
    unsigned lds1 = (unsigned)(uintptr_t)&btile[1][0];

#define ISSUE_TILE(k0, ldsb)                                                    \
    if (tid < 128) {                                                            \
        unsigned dst = (ldsb) + (unsigned)(ldrow * 32 + ldcol);                 \
        const signed char* src = Wp + (size_t)((k0) + ldrow) * N + ldcol;       \
        asm volatile("global_load_async_to_lds_b128 %0, %1, off"                \
                     :: "v"(dst), "v"(src) : "memory");                         \
    }

    v8i acc0, acc1;
#pragma unroll
    for (int i = 0; i < 8; ++i) { acc0[i] = 0; acc1[i] = 0; }

    ISSUE_TILE(0, lds0);
    asm volatile("s_wait_asynccnt 0x0" ::: "memory");
    __syncthreads();

    int buf = 0;
    for (int k0 = 0; k0 < Kd; k0 += 64, buf ^= 1) {
        if (k0 + 64 < Kd) { ISSUE_TILE(k0 + 64, buf ? lds0 : lds1); }
        if (k0 + 128 < Kd) __builtin_prefetch(Ap + k0 + 128, 0, 3);

        v8i af;
#pragma unroll
        for (int v = 0; v < 8; ++v) {
            int ko = ((v & 1) << 2) + (((v >> 1) & 1) << 4) + ((v >> 2) << 5);
            af[v] = *(const int*)(Ap + k0 + ko);
        }
        const signed char* bt = &btile[buf][0];
        v8i bf0, bf1;
#pragma unroll
        for (int v = 0; v < 8; ++v) {
            int kk = ((v & 1) << 2) + (((v >> 1) & 1) << 4) + ((v >> 2) << 5) + (half << 3);
            const unsigned char* bp = (const unsigned char*)(bt + kk * 32 + row);
            bf0[v] = (int)((unsigned)bp[0] | ((unsigned)bp[32] << 8) | ((unsigned)bp[64] << 16) | ((unsigned)bp[96] << 24));
            bf1[v] = (int)((unsigned)bp[16] | ((unsigned)bp[48] << 8) | ((unsigned)bp[80] << 16) | ((unsigned)bp[112] << 24));
        }
        acc0 = __builtin_amdgcn_wmma_i32_16x16x64_iu8(true, af, true, bf0, acc0, false, false);
        acc1 = __builtin_amdgcn_wmma_i32_16x16x64_iu8(true, af, true, bf1, acc1, false, false);

        asm volatile("s_wait_asynccnt 0x0" ::: "memory");
        __syncthreads();
    }

    float ws = *wscale;
    int orow = tm * 16 + 8 * half;
    int oc   = blkN * 32 + row;
#pragma unroll
    for (int r = 0; r < 8; ++r) {
        float as = ascale[orow + r] * ws;
        C[(size_t)(orow + r) * N + oc]      = (float)acc0[r] * as;
        C[(size_t)(orow + r) * N + oc + 16] = (float)acc1[r] * as;
    }
#undef ISSUE_TILE
}

__global__ void __launch_bounds__(256) lnmod_kernel(const float* __restrict__ X,
                                                    const float* __restrict__ mods,
                                                    int shIdx, int scIdx,
                                                    float* __restrict__ Y) {
    __shared__ float red[256];
    int m = blockIdx.x, tid = threadIdx.x;
    const float* x = X + (size_t)m * DIMD;
    float s1 = 0.f, s2 = 0.f;
    for (int d = tid; d < DIMD; d += 256) { float v = x[d]; s1 += v; s2 += v * v; }
    float mean = block_red_sum(s1, red) / (float)DIMD;
    float msq  = block_red_sum(s2, red) / (float)DIMD;
    float var  = fmaxf(msq - mean * mean, 0.f);
    float r = rsqrtf(var + EPS_LN);
    int b = m / SEQ;
    const float* sh = mods + (size_t)b * 12 * DIMD + (size_t)shIdx * DIMD;
    const float* sc = mods + (size_t)b * 12 * DIMD + (size_t)scIdx * DIMD;
    for (int d = tid; d < DIMD; d += 256)
        Y[(size_t)m * DIMD + d] = (x[d] - mean) * r * (1.0f + sc[d]) + sh[d];
}

__global__ void __launch_bounds__(256) hgrn_ew_kernel(float* __restrict__ ib,
                                                      float* __restrict__ fb, size_t n) {
    size_t i = (size_t)blockIdx.x * 256 + threadIdx.x;
    if (i >= n) return;
    float f = sigm(fb[i]);
    float iv = ib[i];
    ib[i] = iv * sigm(iv) * (1.0f - f);
    fb[i] = f;
}

__global__ void __launch_bounds__(256) scan_chunk_kernel(const float* __restrict__ ib,
                                                         const float* __restrict__ fb,
                                                         float* __restrict__ ca,
                                                         float* __restrict__ cb) {
    int idx = blockIdx.x * 256 + threadIdx.x;
    if (idx >= BATCH * NCH * DIMD) return;
    int d = idx % DIMD, ch = (idx / DIMD) % NCH, b = idx / (DIMD * NCH);
    size_t base = ((size_t)b * SEQ + (size_t)ch * CLEN) * DIMD + d;
    float a = 1.f, bb = 0.f;
    for (int t = 0; t < CLEN; ++t) {
        float f = fb[base + (size_t)t * DIMD];
        float i = ib[base + (size_t)t * DIMD];
        a *= f; bb = f * bb + i;
    }
    ca[idx] = a; cb[idx] = bb;
}
__global__ void __launch_bounds__(256) scan_combine_kernel(const float* __restrict__ ca,
                                                           const float* __restrict__ cb,
                                                           float* __restrict__ h0) {
    int idx = blockIdx.x * 256 + threadIdx.x;
    if (idx >= BATCH * DIMD) return;
    int d = idx % DIMD, b = idx / DIMD;
    float h = 0.f;
    for (int ch = 0; ch < NCH; ++ch) {
        size_t o = ((size_t)b * NCH + ch) * DIMD + d;
        h0[o] = h;
        h = ca[o] * h + cb[o];
    }
}
__global__ void __launch_bounds__(256) scan_apply_kernel(const float* __restrict__ ib,
                                                         const float* __restrict__ fb,
                                                         const float* __restrict__ h0,
                                                         float* __restrict__ hb) {
    int idx = blockIdx.x * 256 + threadIdx.x;
    if (idx >= BATCH * NCH * DIMD) return;
    int d = idx % DIMD, ch = (idx / DIMD) % NCH, b = idx / (DIMD * NCH);
    size_t base = ((size_t)b * SEQ + (size_t)ch * CLEN) * DIMD + d;
    float h = h0[idx];
    for (int t = 0; t < CLEN; ++t) {
        size_t o = base + (size_t)t * DIMD;
        h = fb[o] * h + ib[o];
        hb[o] = h;
    }
}

__global__ void __launch_bounds__(256) gate_kernel(const float* __restrict__ h,
                                                   const float* __restrict__ g,
                                                   const float* __restrict__ gn,
                                                   float* __restrict__ o) {
    __shared__ float red[256];
    int m = blockIdx.x, tid = threadIdx.x;
    const float* hr = h + (size_t)m * DIMD;
    const float* gr = g + (size_t)m * DIMD;
    float ss = 0.f;
    for (int d = tid; d < DIMD; d += 256) { float v = hr[d]; ss += v * v; }
    float tot = block_red_sum(ss, red);
    float r = rsqrtf(tot / (float)DIMD + EPS_BIT);
    for (int d = tid; d < DIMD; d += 256) {
        float gv = gr[d];
        o[(size_t)m * DIMD + d] = hr[d] * r * gn[d] * gv * sigm(gv);
    }
}

__global__ void __launch_bounds__(256) swiglu_kernel(const float* __restrict__ Y2,
                                                     float* __restrict__ Z,
                                                     int halfN, int ld, size_t n) {
    size_t idx = (size_t)blockIdx.x * 256 + threadIdx.x;
    if (idx >= n) return;
    size_t m = idx / halfN; int j = (int)(idx % halfN);
    float a = Y2[m * ld + j], b = Y2[m * ld + halfN + j];
    Z[idx] = a * sigm(a) * b;
}

__global__ void __launch_bounds__(256) silu_kernel(const float* __restrict__ x,
                                                   float* __restrict__ y, int n) {
    int i = blockIdx.x * 256 + threadIdx.x;
    if (i < n) { float v = x[i]; y[i] = v * sigm(v); }
}

__global__ void __launch_bounds__(256) resid_kernel(float* __restrict__ R,
                                                    const float* __restrict__ y,
                                                    const float* __restrict__ mods,
                                                    int gidx, size_t n) {
    size_t idx = (size_t)blockIdx.x * 256 + threadIdx.x;
    if (idx >= n) return;
    int d = (int)(idx % DIMD);
    int b = (int)(idx / ((size_t)SEQ * DIMD));
    float g = (gidx >= 0) ? mods[(size_t)b * 12 * DIMD + (size_t)gidx * DIMD + d] : 1.0f;
    R[idx] += g * y[idx];
}

__global__ void __launch_bounds__(256) gemm_f32_kernel(const float* __restrict__ A,
                                                       const float* __restrict__ W,
                                                       const float* __restrict__ bias,
                                                       float* __restrict__ C, int N, int Kd) {
    int n = blockIdx.x * 256 + threadIdx.x;
    int m = blockIdx.y;
    if (n >= N) return;
    const float* a = A + (size_t)m * Kd;
    float acc = bias ? bias[n] : 0.f;
    for (int k = 0; k < Kd; ++k) acc += a[k] * W[(size_t)k * N + n];
    C[(size_t)m * N + n] = acc;
}

__global__ void __launch_bounds__(256) rms_row_kernel(const float* __restrict__ X,
                                                      const float* __restrict__ w,
                                                      float* __restrict__ Y, int Kd, float eps) {
    __shared__ float red[256];
    int m = blockIdx.x, tid = threadIdx.x;
    const float* x = X + (size_t)m * Kd;
    float ss = 0.f;
    for (int k = tid; k < Kd; k += 256) { float v = x[k]; ss += v * v; }
    float tot = block_red_sum(ss, red);
    float r = rsqrtf(tot / (float)Kd + eps);
    for (int k = tid; k < Kd; k += 256) Y[(size_t)m * Kd + k] = x[k] * r * w[k];
}

__global__ void __launch_bounds__(128) flash_attn_kernel(const float* __restrict__ Q,
                                                         const float* __restrict__ Km,
                                                         const float* __restrict__ Vm,
                                                         float* __restrict__ O) {
    const int ldq = DIMD;
    int gw   = (blockIdx.x * blockDim.x + threadIdx.x) >> 5;
    int lane = threadIdx.x & 31;
    const int T16 = SEQ / 16;
    if (gw >= BATCH * HEADS * T16) return;
    int b   = gw / (HEADS * T16);
    int rem = gw % (HEADS * T16);
    int h   = rem / T16;
    int q0  = (rem % T16) * 16;
    int row = lane & 15, half = lane >> 4;

    __shared__ _Float16 pbuf[4][16 * 32];
    _Float16* P = pbuf[threadIdx.x >> 5];

    const float* qp = Q + ((size_t)(b * SEQ + q0 + row)) * ldq + h * HD;
    v16h qa0, qa1;
#pragma unroll
    for (int v = 0; v < 8; ++v) {
        int kp = ((v & 3) << 1) + ((v >> 2) << 4) + (half << 3);
        qa0[2 * v]     = (_Float16)qp[kp];
        qa0[2 * v + 1] = (_Float16)qp[kp + 1];
        qa1[2 * v]     = (_Float16)qp[32 + kp];
        qa1[2 * v + 1] = (_Float16)qp[32 + kp + 1];
    }
    const float* Kb = Km + (size_t)b * SEQ * ldq + h * HD;
    const float* Vb = Vm + (size_t)b * SEQ * ldq + h * HD;

    float mrow[8], lrow[8];
    v8f oacc[4];
#pragma unroll
    for (int r = 0; r < 8; ++r) { mrow[r] = -1e30f; lrow[r] = 0.f; }
#pragma unroll
    for (int j = 0; j < 4; ++j)
#pragma unroll
        for (int r = 0; r < 8; ++r) oacc[j][r] = 0.f;

    const float sscale = 0.125f;

    for (int kv0 = 0; kv0 < SEQ; kv0 += 32) {
        v8f s0, s1;
#pragma unroll
        for (int r = 0; r < 8; ++r) { s0[r] = 0.f; s1[r] = 0.f; }
        {
            v16h bk;
#pragma unroll
            for (int v = 0; v < 8; ++v) { int kr = kv0 + row; int kp = ((v & 3) << 1) + ((v >> 2) << 4) + (half << 3);
                bk[2 * v]     = (_Float16)Kb[(size_t)kr * ldq + kp];
                bk[2 * v + 1] = (_Float16)Kb[(size_t)kr * ldq + kp + 1]; }
            s0 = __builtin_amdgcn_wmma_f32_16x16x32_f16(false, qa0, false, bk, (short)0, s0, false, false);
#pragma unroll
            for (int v = 0; v < 8; ++v) { int kr = kv0 + row; int kp = ((v & 3) << 1) + ((v >> 2) << 4) + (half << 3);
                bk[2 * v]     = (_Float16)Kb[(size_t)kr * ldq + 32 + kp];
                bk[2 * v + 1] = (_Float16)Kb[(size_t)kr * ldq + 32 + kp + 1]; }
            s0 = __builtin_amdgcn_wmma_f32_16x16x32_f16(false, qa1, false, bk, (short)0, s0, false, false);
#pragma unroll
            for (int v = 0; v < 8; ++v) { int kr = kv0 + 16 + row; int kp = ((v & 3) << 1) + ((v >> 2) << 4) + (half << 3);
                bk[2 * v]     = (_Float16)Kb[(size_t)kr * ldq + kp];
                bk[2 * v + 1] = (_Float16)Kb[(size_t)kr * ldq + kp + 1]; }
            s1 = __builtin_amdgcn_wmma_f32_16x16x32_f16(false, qa0, false, bk, (short)0, s1, false, false);
#pragma unroll
            for (int v = 0; v < 8; ++v) { int kr = kv0 + 16 + row; int kp = ((v & 3) << 1) + ((v >> 2) << 4) + (half << 3);
                bk[2 * v]     = (_Float16)Kb[(size_t)kr * ldq + 32 + kp];
                bk[2 * v + 1] = (_Float16)Kb[(size_t)kr * ldq + 32 + kp + 1]; }
            s1 = __builtin_amdgcn_wmma_f32_16x16x32_f16(false, qa1, false, bk, (short)0, s1, false, false);
        }
#pragma unroll
        for (int r = 0; r < 8; ++r) { s0[r] *= sscale; s1[r] *= sscale; }

        float rmax[8];
#pragma unroll
        for (int r = 0; r < 8; ++r) rmax[r] = fmaxf(s0[r], s1[r]);
#pragma unroll
        for (int off = 1; off < 16; off <<= 1)
#pragma unroll
            for (int r = 0; r < 8; ++r) rmax[r] = fmaxf(rmax[r], __shfl_xor(rmax[r], off, 32));

#pragma unroll
        for (int r = 0; r < 8; ++r) {
            float mn = fmaxf(mrow[r], rmax[r]);
            float c  = __expf(mrow[r] - mn);
            float p0 = __expf(s0[r] - mn);
            float p1 = __expf(s1[r] - mn);
            float rs = p0 + p1;
            for (int off = 1; off < 16; off <<= 1) rs += __shfl_xor(rs, off, 32);
            lrow[r] = lrow[r] * c + rs;
            mrow[r] = mn;
            oacc[0][r] *= c; oacc[1][r] *= c; oacc[2][r] *= c; oacc[3][r] *= c;
            P[(r + 8 * half) * 32 + row]      = (_Float16)p0;
            P[(r + 8 * half) * 32 + 16 + row] = (_Float16)p1;
        }

        v16h pf;
#pragma unroll
        for (int v = 0; v < 8; ++v) {
            int kp = ((v & 3) << 1) + ((v >> 2) << 4) + (half << 3);
            pf[2 * v]     = P[row * 32 + kp];
            pf[2 * v + 1] = P[row * 32 + kp + 1];
        }
#pragma unroll
        for (int j = 0; j < 4; ++j) {
            v16h vf;
#pragma unroll
            for (int v = 0; v < 8; ++v) {
                int kp = ((v & 3) << 1) + ((v >> 2) << 4) + (half << 3);
                const float* vp = Vb + (size_t)(kv0 + kp) * ldq + j * 16 + row;
                vf[2 * v] = (_Float16)vp[0]; vf[2 * v + 1] = (_Float16)vp[ldq];
            }
            oacc[j] = __builtin_amdgcn_wmma_f32_16x16x32_f16(false, pf, false, vf, (short)0, oacc[j], false, false);
        }
    }

#pragma unroll
    for (int r = 0; r < 8; ++r) {
        float inv = 1.0f / lrow[r];
        size_t ro = (size_t)(b * SEQ + q0 + r + 8 * half) * ldq + h * HD;
#pragma unroll
        for (int j = 0; j < 4; ++j) O[ro + j * 16 + row] = oacc[j][r] * inv;
    }
}

static inline size_t al256(size_t x) { return (x + 255) & ~(size_t)255; }

extern "C" void kernel_launch(void* const* d_in, const int* in_sizes, int n_in,
                              void* d_out, int out_size, void* d_ws, size_t ws_size,
                              hipStream_t stream) {
    (void)in_sizes; (void)out_size; (void)ws_size;
    if (n_in < 45) return;

    const int IN_X = 0, IN_C = 1, IN_T = 2;
    const int ADA = 3;
    const int AX = 11, AC = 20;
    const int CRP = 29;
    const int MX = 37, MC = 41;

    char* base = (char*)d_ws;
    size_t off = 0;
    auto alloc = [&](size_t bytes) { char* p = base + off; off = al256(off + bytes); return (void*)p; };

    const int DD = DIMD * DIMD;
    int wsrc[16] = { AX+0, AX+1, AX+2, AX+3,  AC+0, AC+1, AC+2, AC+3,
                     CRP+0, CRP+1, CRP+2, CRP+3,  MX+0, MX+2, MC+0, MC+2 };
    int wn[16]   = { DD, DD, DD, DD,  DD, DD, DD, DD,  DD, DD, DD, DD,
                     DIMD*IFULL, IHALF*DIMD, DIMD*IFULL, IHALF*DIMD };
    signed char* WQ8[16];
    for (int i = 0; i < 16; ++i) WQ8[i] = (signed char*)alloc((size_t)wn[i]);

    float* WSUM = (float*)alloc(16 * 4);
    float* WSC  = (float*)alloc(16 * 4);
    float* MODS = (float*)alloc((size_t)BATCH * 12 * DIMD * 4);
    float* TS   = (float*)alloc((size_t)BATCH * DIMD * 4);
    float* H1   = (float*)alloc((size_t)BATCH * DIMD * 4);
    float* H2   = (float*)alloc((size_t)BATCH * IFULL * 4);
    float* Z2   = (float*)alloc((size_t)BATCH * IHALF * 4);
    float* H3   = (float*)alloc((size_t)BATCH * DIMD * 4);
    float* H4   = (float*)alloc((size_t)BATCH * 12 * DIMD * 4);
    float* H5   = (float*)alloc((size_t)BATCH * 12 * DIMD * 4);
    float* XM   = (float*)alloc((size_t)MTOT * DIMD * 4);
    signed char* A8 = (signed char*)alloc((size_t)MTOT * IHALF);
    float* ASC  = (float*)alloc((size_t)MTOT * 4);
    float* BUFA = (float*)alloc((size_t)MTOT * DIMD * 4);
    float* BUFB = (float*)alloc((size_t)MTOT * DIMD * 4);
    float* BUFC = (float*)alloc((size_t)MTOT * DIMD * 4);
    float* BUFD = (float*)alloc((size_t)MTOT * DIMD * 4);
    float* BUF0 = (float*)alloc((size_t)MTOT * IFULL * 4);
    float* BUFZ = (float*)alloc((size_t)MTOT * IHALF * 4);
    float* SCA  = (float*)alloc((size_t)BATCH * NCH * DIMD * 4);
    float* SCB  = (float*)alloc((size_t)BATCH * NCH * DIMD * 4);
    float* SH0  = (float*)alloc((size_t)BATCH * NCH * DIMD * 4);

    const size_t MD = (size_t)MTOT * DIMD;
    float* xres = (float*)d_out;
    float* cres = xres + MD;
    hipMemcpyAsync(xres, d_in[IN_X], MD * 4, hipMemcpyDeviceToDevice, stream);
    hipMemcpyAsync(cres, d_in[IN_C], MD * 4, hipMemcpyDeviceToDevice, stream);

    hipMemsetAsync(WSUM, 0, 16 * 4, stream);
    for (int i = 0; i < 16; ++i) {
        int blocks = (wn[i] + 255) / 256; if (blocks > 2048) blocks = 2048;
        abs_sum_kernel<<<blocks, 256, 0, stream>>>((const float*)d_in[wsrc[i]], wn[i], WSUM + i);
    }
    for (int i = 0; i < 16; ++i)
        wquant_kernel<<<(wn[i] + 255) / 256, 256, 0, stream>>>((const float*)d_in[wsrc[i]],
                                                               WQ8[i], wn[i], WSUM + i, WSC + i);

    silu_kernel<<<(BATCH * DIMD + 255) / 256, 256, 0, stream>>>((const float*)d_in[IN_T], TS, BATCH * DIMD);
    gemm_f32_kernel<<<dim3((DIMD + 255) / 256, BATCH), 256, 0, stream>>>(TS, (const float*)d_in[ADA + 0], nullptr, H1, DIMD, DIMD);
    gemm_f32_kernel<<<dim3((IFULL + 255) / 256, BATCH), 256, 0, stream>>>(H1, (const float*)d_in[ADA + 1], nullptr, H2, IFULL, DIMD);
    swiglu_kernel<<<(BATCH * IHALF + 255) / 256, 256, 0, stream>>>(H2, Z2, IHALF, IFULL, (size_t)BATCH * IHALF);
    gemm_f32_kernel<<<dim3((DIMD + 255) / 256, BATCH), 256, 0, stream>>>(Z2, (const float*)d_in[ADA + 2], nullptr, H3, DIMD, IHALF);
    gemm_f32_kernel<<<dim3((12 * DIMD + 255) / 256, BATCH), 256, 0, stream>>>(H3, (const float*)d_in[ADA + 3], (const float*)d_in[ADA + 4], H4, 12 * DIMD, DIMD);
    rms_row_kernel<<<BATCH, 256, 0, stream>>>(H4, (const float*)d_in[ADA + 5], H5, 12 * DIMD, EPS_LN);
    gemm_f32_kernel<<<dim3((12 * DIMD + 255) / 256, BATCH), 256, 0, stream>>>(H5, (const float*)d_in[ADA + 6], (const float*)d_in[ADA + 7], MODS, 12 * DIMD, 12 * DIMD);

    auto gemm_q = [&](const signed char* a, int qi, float* c, int N, int Kd) {
        int blocks = (MTOT / 128) * (N / 32);
        gemm_iu8_kernel<<<blocks, 256, 0, stream>>>(a, WQ8[qi], c, ASC, WSC + qi, MTOT, N, Kd);
    };
    auto bitlin = [&](const float* x, int gammaIdx, int qi, float* c, int N, int Kd) {
        rmsq_kernel<<<MTOT, 256, 0, stream>>>(x, (const float*)d_in[gammaIdx], A8, ASC, Kd);
        gemm_q(A8, qi, c, N, Kd);
    };

    const int nEW   = (int)((MD + 255) / 256);
    const int nSCAN = (BATCH * NCH * DIMD + 255) / 256;

    auto run_hgrn = [&](float* R, int qbase, int pbase, int g0) {
        lnmod_kernel<<<MTOT, 256, 0, stream>>>(R, MODS, g0, g0 + 1, XM);
        bitlin(XM, pbase + 4, qbase + 0, BUFA, DIMD, DIMD);
        bitlin(XM, pbase + 5, qbase + 1, BUFB, DIMD, DIMD);
        bitlin(XM, pbase + 6, qbase + 2, BUFD, DIMD, DIMD);
        hgrn_ew_kernel<<<nEW, 256, 0, stream>>>(BUFA, BUFB, MD);
        scan_chunk_kernel<<<nSCAN, 256, 0, stream>>>(BUFA, BUFB, SCA, SCB);
        scan_combine_kernel<<<(BATCH * DIMD + 255) / 256, 256, 0, stream>>>(SCA, SCB, SH0);
        scan_apply_kernel<<<nSCAN, 256, 0, stream>>>(BUFA, BUFB, SH0, BUFC);
        gate_kernel<<<MTOT, 256, 0, stream>>>(BUFC, BUFD, (const float*)d_in[pbase + 8], XM);
        bitlin(XM, pbase + 7, qbase + 3, BUFA, DIMD, DIMD);
        resid_kernel<<<nEW, 256, 0, stream>>>(R, BUFA, MODS, g0 + 2, MD);
    };

    auto run_mlp = [&](float* R, int qg, int qd, int pbase, int g0) {
        lnmod_kernel<<<MTOT, 256, 0, stream>>>(R, MODS, g0, g0 + 1, XM);
        bitlin(XM, pbase + 1, qg, BUF0, IFULL, DIMD);
        swiglu_kernel<<<(int)(((size_t)MTOT * IHALF + 255) / 256), 256, 0, stream>>>(BUF0, BUFZ, IHALF, IFULL, (size_t)MTOT * IHALF);
        bitlin(BUFZ, pbase + 3, qd, BUFA, DIMD, IHALF);
        resid_kernel<<<nEW, 256, 0, stream>>>(R, BUFA, MODS, g0 + 2, MD);
    };

    run_hgrn(xres, 0, AX, 0);
    run_mlp (xres, 12, 13, MX, 3);
    run_hgrn(cres, 4, AC, 6);
    run_mlp (cres, 14, 15, MC, 9);

    bitlin(xres, CRP + 4, 8,  BUFA, DIMD, DIMD);
    bitlin(cres, CRP + 5, 9,  BUFB, DIMD, DIMD);
    bitlin(cres, CRP + 6, 10, BUFC, DIMD, DIMD);
    {
        int waves = BATCH * HEADS * (SEQ / 16);
        flash_attn_kernel<<<(waves * 32 + 127) / 128, 128, 0, stream>>>(BUFA, BUFB, BUFC, BUFD);
    }
    bitlin(BUFD, CRP + 7, 11, XM, DIMD, DIMD);
    resid_kernel<<<nEW, 256, 0, stream>>>(xres, XM, MODS, -1, MD);
}
